// VectorQuantizer_WeightCodebook_16329465659950
// MI455X (gfx1250) — matrix operand order repaired and checked by reading, not compiled, not run on hardware
//
#include <hip/hip_runtime.h>
#include <stdint.h>

// Problem constants (from reference)
#define EDIM   512
#define NE     1024
#define NTOT   (16 * 64 * 64)          // 65536 latent vectors
#define ZQ_ELEMS (16 * 512 * 64 * 64)  // 33554432
#define LOSS_OFF ZQ_ELEMS
#define IDX_OFF  (ZQ_ELEMS + 1)

#define ROWS_PER_BLOCK 128
#define LDK 520                        // padded K-stride (halves): 1040B, 16B aligned, bank-spread

typedef __attribute__((ext_vector_type(16))) __bf16 v16bf;
typedef __attribute__((ext_vector_type(8)))  float  v8f;

union U16x { uint4 q[2]; v16bf v; };

__device__ __forceinline__ unsigned short f32_to_bf16(float f) {
  union { float f; unsigned int u; } x; x.f = f;
  unsigned int u = x.u;
  unsigned int r = u + 0x7FFFu + ((u >> 16) & 1u);   // round-to-nearest-even
  return (unsigned short)(r >> 16);
}

// ---------------------------------------------------------------------------
// Prep: codebook f32 -> bf16 (workspace), per-entry squared norms, zero loss.
// ---------------------------------------------------------------------------
__global__ void vq_prep(const float* __restrict__ cb,
                        unsigned short* __restrict__ cbB,
                        float* __restrict__ cn,
                        float* __restrict__ lossPtr) {
  const int j = blockIdx.x;          // codebook entry
  const int t = threadIdx.x;         // 128 threads
  if (j == 0 && t == 0) *lossPtr = 0.0f;
  float s = 0.0f;
  for (int c = t; c < EDIM; c += 128) {
    float v = cb[j * EDIM + c];
    cbB[j * EDIM + c] = f32_to_bf16(v);
    s += v * v;
  }
  for (int m = 16; m > 0; m >>= 1) s += __shfl_down(s, m, 32);
  __shared__ float red[4];
  const int lane = t & 31, wv = t >> 5;
  if (lane == 0) red[wv] = s;
  __syncthreads();
  if (t == 0) cn[j] = red[0] + red[1] + red[2] + red[3];
}

// ---------------------------------------------------------------------------
// Main: bf16 WMMA distance GEMM + argmin + gather + fused loss.
// 512 blocks x 256 threads (8 waves). Each block: 128 z-rows vs all 1024 codes.
// ---------------------------------------------------------------------------
__global__ void vq_main(const float* __restrict__ z,
                        const float* __restrict__ cbF,
                        const unsigned short* __restrict__ cbB,
                        const float* __restrict__ cn,
                        float* __restrict__ zq,
                        float* __restrict__ lossPtr,
                        float* __restrict__ idxOut) {
  extern __shared__ unsigned char smemRaw[];
  unsigned short* sA   = (unsigned short*)smemRaw;                 // 128*LDK halves
  int*            sIdx = (int*)(smemRaw + ROWS_PER_BLOCK * LDK * 2);
  float*          sRed = (float*)(sIdx + ROWS_PER_BLOCK);

  const int t  = threadIdx.x;                 // 256
  const int n0 = blockIdx.x * ROWS_PER_BLOCK; // first flat row (b,h,w)
  const int b  = n0 >> 12;                    // 4096 rows per batch image
  const int h0 = (n0 & 4095) >> 6;            // block spans h0, h0+1 (full w)

  // --- Stage A tile: 128 rows x 512 dims, transpose from (B,C,H,W), f32->bf16.
  // For fixed c, the 128 rows are 128 contiguous floats -> fully coalesced.
  {
    const int  i      = t & 127;
    const int  c0     = t >> 7;
    const long rowOff = (long)h0 * 64 + i;
    for (int c = c0; c < EDIM; c += 2) {
      float v = z[((long)(b * EDIM + c) << 12) + rowOff];
      sA[i * LDK + c] = f32_to_bf16(v);
    }
  }
  __syncthreads();

  const int lane = t & 31;
  const int wv   = t >> 5;        // wave id = M-tile (16 rows each)
  const int r15  = lane & 15;
  const int hi   = lane >> 4;     // half-wave select

  // ISA A layout (16-bit 16x32): lo lanes K{0..7,16..23}, hi lanes K{8..15,24..31}
  const unsigned short* aBase = sA + (wv * 16 + r15) * LDK + hi * 8;

  float bestD[8];
  int   bestI[8];
#pragma unroll
  for (int r = 0; r < 8; ++r) { bestD[r] = 3.4e38f; bestI[r] = 0; }

  for (int nt = 0; nt < 64; ++nt) {
    const int N = nt * 16 + r15;  // this lane's codebook column
    // B layout matches A: lo lanes K{0..7,16..23}, hi lanes K{8..15,24..31}
    const unsigned short* bp = cbB + N * EDIM + hi * 8;
    if (nt < 63) __builtin_prefetch(bp + 16 * EDIM, 0, 1);  // next tile's B row

    v8f acc;
#pragma unroll
    for (int r = 0; r < 8; ++r) acc[r] = 0.0f;

#pragma unroll 4
    for (int kk = 0; kk < EDIM; kk += 32) {
      U16x a, bb;
      const unsigned short* ap = aBase + kk;
      a.q[0]  = *(const uint4*)(ap);
      a.q[1]  = *(const uint4*)(ap + 16);
      bb.q[0] = *(const uint4*)(bp + kk);
      bb.q[1] = *(const uint4*)(bp + kk + 16);
      acc = __builtin_amdgcn_wmma_f32_16x16x32_bf16(
          false, a.v, false, bb.v, (short)0, acc, false, false);
    }

    const float c2 = cn[N];
#pragma unroll
    for (int r = 0; r < 8; ++r) {           // d = ||c||^2 - 2*z.c  (||z||^2 const)
      float d = c2 - 2.0f * acc[r];
      if (d < bestD[r]) { bestD[r] = d; bestI[r] = N; }
    }
  }

  // Cross-lane argmin within each 16-lane half (ties -> lower index, jnp.argmin)
#pragma unroll
  for (int m = 1; m < 16; m <<= 1) {
#pragma unroll
    for (int r = 0; r < 8; ++r) {
      float od = __shfl_xor(bestD[r], m, 32);
      int   oi = __shfl_xor(bestI[r], m, 32);
      if (od < bestD[r] || (od == bestD[r] && oi < bestI[r])) {
        bestD[r] = od; bestI[r] = oi;
      }
    }
  }
  if (lane == 0 || lane == 16) {
#pragma unroll
    for (int r = 0; r < 8; ++r) {
      const int row = wv * 16 + hi * 8 + r;   // C/D layout: M = r + 8*hi
      sIdx[row] = bestI[r];
      idxOut[n0 + row] = (float)bestI[r];
    }
  }
  __syncthreads();

  // --- Gather z_q = codebook[idx], write (B,C,H,W) output, fused loss.
  float lsum = 0.0f;
  {
    const int   i      = t & 127;
    const int   c0     = t >> 7;
    const long  rowOff = (long)h0 * 64 + i;
    const int   j      = sIdx[i];
    const float* cbRow = cbF + (long)j * EDIM;
    for (int c = c0; c < EDIM; c += 2) {
      const long addr = ((long)(b * EDIM + c) << 12) + rowOff;
      const float q = cbRow[c];
      const float d = q - z[addr];
      lsum += d * d;
      zq[addr] = q;      // straight-through: z_q_out == codebook[idx]
    }
  }
  for (int m = 16; m > 0; m >>= 1) lsum += __shfl_down(lsum, m, 32);
  if (lane == 0) sRed[wv] = lsum;
  __syncthreads();
  if (t == 0) {
    float s = 0.0f;
#pragma unroll
    for (int w = 0; w < 8; ++w) s += sRed[w];
    // codebook_loss = q_loss + beta*e_loss = (1 + 0.25) * mean((z_q - z)^2)
    atomicAdd(lossPtr, s * (1.25f / (float)ZQ_ELEMS));
  }
}

// ---------------------------------------------------------------------------
extern "C" void kernel_launch(void* const* d_in, const int* in_sizes, int n_in,
                              void* d_out, int out_size, void* d_ws, size_t ws_size,
                              hipStream_t stream) {
  const float* z  = (const float*)d_in[0];       // (16,512,64,64) f32
  const float* cb = (const float*)d_in[1];       // (1024,512) f32

  unsigned short* cbB = (unsigned short*)d_ws;                     // 1 MB bf16 codebook
  float* cnorm = (float*)((char*)d_ws + (size_t)NE * EDIM * 2);    // 4 KB norms

  float* out     = (float*)d_out;
  float* zq      = out;                 // 33 554 432 floats
  float* lossPtr = out + LOSS_OFF;      // 1 float
  float* idxOut  = out + IDX_OFF;       // 65 536 floats (indices as float)

  vq_prep<<<NE, 128, 0, stream>>>(cb, cbB, cnorm, lossPtr);

  const size_t shmem = (size_t)ROWS_PER_BLOCK * LDK * 2   // A tile (bf16)
                     + (size_t)ROWS_PER_BLOCK * 4         // idx
                     + 8 * 4;                             // reduction
  vq_main<<<NTOT / ROWS_PER_BLOCK, 256, shmem, stream>>>(
      z, cb, cbB, cnorm, zq, lossPtr, idxOut);
}
